// OliVe_Linear_45535243273022
// MI455X (gfx1250) — matrix operand order repaired and checked by reading, not compiled, not run on hardware
//
#include <hip/hip_runtime.h>

typedef __attribute__((ext_vector_type(16))) _Float16 v16h;
typedef __attribute__((ext_vector_type(8)))  _Float16 v8h;
typedef __attribute__((ext_vector_type(8)))  float    v8f;
typedef __attribute__((ext_vector_type(4)))  _Float16 v4h;
typedef __attribute__((ext_vector_type(4)))  float    v4f;

#define GROUP 128
#define MAX_Q 28.0f

constexpr int Mdim = 2048, Ndim = 2048, Kdim = 2048;

// ---------------------------------------------------------------------------
// Quantization: one wave (32 lanes) handles one contiguous group of 128 floats.
// lane l holds elements [4l, 4l+4): two aligned pairs -> victim masking local.
// ---------------------------------------------------------------------------
__device__ __forceinline__ float nearest_grid(float u) {
    // int grid: multiples of 4 in [-28, 28]
    float qi = fminf(fmaxf(rintf(u * 0.25f) * 4.0f, -MAX_Q), MAX_Q);
    // outlier grid (positive): 48 64 96 128 192 256 384; midpoints 56 80 112 160 224 320
    float au = fabsf(u);
    float ao;
    if (au < 160.0f) {
        if (au < 80.0f)  ao = (au < 56.0f)  ? 48.0f  : 64.0f;
        else             ao = (au < 112.0f) ? 96.0f  : 128.0f;
    } else {
        if (au < 320.0f) ao = (au < 224.0f) ? 192.0f : 256.0f;
        else             ao = 384.0f;
    }
    float qo = copysignf(ao, u);
    // tie -> int grid (it comes first in the merged-grid argmin)
    return (fabsf(u - qi) <= fabsf(u - qo)) ? qi : qo;
}

__global__ __launch_bounds__(256)
void olive_quant_kernel(const float* __restrict__ in, _Float16* __restrict__ out,
                        int n_groups, float alpha) {
    const int lane = threadIdx.x & 31;
    const int wave = threadIdx.x >> 5;
    const int group = blockIdx.x * 8 + wave;
    if (group >= n_groups) return;

    const float* g = in + (size_t)group * GROUP;
    v4f v = *(const v4f*)(g + lane * 4);

    float s  = v.x + v.y + v.z + v.w;
    float ss = v.x * v.x + v.y * v.y + v.z * v.z + v.w * v.w;
    #pragma unroll
    for (int off = 16; off >= 1; off >>= 1) {
        s  += __shfl_xor(s,  off, 32);
        ss += __shfl_xor(ss, off, 32);
    }
    float mean = s * (1.0f / 128.0f);
    float var  = fmaxf((ss - 128.0f * mean * mean) * (1.0f / 127.0f), 0.0f);
    float sd   = sqrtf(var);
    float normal_max = fmaxf(fabsf(mean + 3.0f * sd), fabsf(mean - 3.0f * sd));
    float scale = normal_max * alpha * (1.0f / MAX_Q);
    float inv   = (scale > 0.0f) ? (1.0f / scale) : 0.0f;

    float q0 = nearest_grid(v.x * inv);
    float q1 = nearest_grid(v.y * inv);
    float q2 = nearest_grid(v.z * inv);
    float q3 = nearest_grid(v.w * inv);

    // victim masking within each aligned pair
    if (fabsf(q0) > 32.0f)      q1 = 0.0f;
    else if (fabsf(q1) > 32.0f) q0 = 0.0f;
    if (fabsf(q2) > 32.0f)      q3 = 0.0f;
    else if (fabsf(q3) > 32.0f) q2 = 0.0f;

    v4h d;
    d.x = (_Float16)(q0 * scale);
    d.y = (_Float16)(q1 * scale);
    d.z = (_Float16)(q2 * scale);
    d.w = (_Float16)(q3 * scale);
    *(v4h*)(out + (size_t)group * GROUP + lane * 4) = d;
}

// ---------------------------------------------------------------------------
// WMMA GEMM: C[M,N] = Xq[M,K] * Wq[N,K]^T, fp16 in, fp32 accumulate/out.
// Block = 256 threads (8 waves), tile 128x128, BK=32, double-buffered LDS
// filled by GLOBAL_LOAD_ASYNC_TO_LDS_B128 (ASYNCcnt-tracked, CDNA5 path).
// Wave grid 4x2; each wave -> 32x64 = 2x4 subtiles of 16x16.
// ---------------------------------------------------------------------------
#define BM 128
#define BN 128
#define BK 32
#define LDT 40                       // padded LDS row stride in halfs
constexpr int STAGE = BM * LDT;      // halfs per A (or B) stage

__device__ __forceinline__ void async_b128(unsigned lds_byte_addr,
                                           const _Float16* gptr) {
    // LDS byte address in the first operand, 64-bit global address in the second
    asm volatile("global_load_async_to_lds_b128 %0, %1, off"
                 :: "v"(lds_byte_addr), "v"(gptr)
                 : "memory");
}

__global__ __launch_bounds__(256)
void olive_wmma_gemm(const _Float16* __restrict__ Xq, const _Float16* __restrict__ Wq,
                     float* __restrict__ C) {
    // [ A stage0 | A stage1 | B stage0 | B stage1 ]
    __shared__ _Float16 smem[4 * STAGE];
    const unsigned smem_base = (unsigned)(size_t)(void*)&smem[0];

    const int tid  = threadIdx.x;
    const int lane = tid & 31;
    const int wid  = tid >> 5;     // 0..7
    const int wm   = wid >> 1;     // 0..3  -> 32-row band
    const int wn   = wid & 1;      // 0..1  -> 64-col band

    const int block_row = blockIdx.y * BM;
    const int block_col = blockIdx.x * BN;

    v8f acc[2][4];
    #pragma unroll
    for (int mi = 0; mi < 2; ++mi)
        #pragma unroll
        for (int ni = 0; ni < 4; ++ni)
            acc[mi][ni] = (v8f){0.f,0.f,0.f,0.f,0.f,0.f,0.f,0.f};

    // staging: thread t owns 32 bytes (16 halfs) of row (t>>1), segment (t&1)
    const int lr  = tid >> 1;
    const int seg = (tid & 1) * 16;
    const _Float16* gA = Xq + (size_t)(block_row + lr) * Kdim + seg;
    const _Float16* gB = Wq + (size_t)(block_col + lr) * Kdim + seg;
    const unsigned ldsA = smem_base + 2u * (unsigned)(lr * LDT + seg);
    const unsigned ldsB = ldsA + 2u * 2u * (unsigned)STAGE;

    const int half_sel = lane >> 4;   // 0: K-low half group, 1: K-high
    const int lrow     = lane & 15;

    constexpr int NSTEP = Kdim / BK;  // 64

    // prologue: async-stage tile 0 into buffer 0 (4 ops -> ASYNCcnt += 4)
    {
        async_b128(ldsA,      gA);
        async_b128(ldsA + 16, gA + 8);
        async_b128(ldsB,      gB);
        async_b128(ldsB + 16, gB + 8);
    }

    for (int i = 0; i < NSTEP; ++i) {
        const int cur = i & 1;
        if (i + 1 < NSTEP) {
            // prefetch tile i+1 into the other stage
            const int nxt = cur ^ 1;
            const unsigned dA = ldsA + 2u * (unsigned)(nxt * STAGE);
            const unsigned dB = ldsB + 2u * (unsigned)(nxt * STAGE);
            const _Float16* pA = gA + (size_t)(i + 1) * BK;
            const _Float16* pB = gB + (size_t)(i + 1) * BK;
            async_b128(dA,      pA);
            async_b128(dA + 16, pA + 8);
            async_b128(dB,      pB);
            async_b128(dB + 16, pB + 8);
            // tile i's 4 ops complete once only the 4 just-issued remain
            asm volatile("s_wait_asynccnt 0x4" ::: "memory");
        } else {
            asm volatile("s_wait_asynccnt 0x0" ::: "memory");
        }
        __syncthreads();   // tile i visible to every wave

        const _Float16* As = smem + cur * STAGE;
        const _Float16* Bs = smem + (2 + cur) * STAGE;

        // A fragment (16x32 f16): lanes 0-15 (row m): K 0..7 + 16..23;
        // lanes 16-31: K 8..15 + 24..31.
        v16h afrag[2];
        #pragma unroll
        for (int mi = 0; mi < 2; ++mi) {
            const _Float16* base = As + (wm * 32 + mi * 16 + lrow) * LDT;
            v8h lo = *(const v8h*)(base + half_sel * 8);
            v8h hi = *(const v8h*)(base + 16 + half_sel * 8);
            afrag[mi] = __builtin_shufflevector(lo, hi,
                0,1,2,3,4,5,6,7,8,9,10,11,12,13,14,15);
        }
        // B fragment (32x16 f16): lanes 0-15 col n K 0..7 + 16..23; lanes 16-31 K 8..15 + 24..31.
        v16h bfrag[4];
        #pragma unroll
        for (int ni = 0; ni < 4; ++ni) {
            const _Float16* base = Bs + (wn * 64 + ni * 16 + lrow) * LDT;
            v8h lo = *(const v8h*)(base + half_sel * 8);
            v8h hi = *(const v8h*)(base + 16 + half_sel * 8);
            bfrag[ni] = __builtin_shufflevector(lo, hi,
                0,1,2,3,4,5,6,7,8,9,10,11,12,13,14,15);
        }
        __syncthreads();   // all waves done reading stage `cur` before it is re-filled

        #pragma unroll
        for (int mi = 0; mi < 2; ++mi)
            #pragma unroll
            for (int ni = 0; ni < 4; ++ni)
                acc[mi][ni] = __builtin_amdgcn_wmma_f32_16x16x32_f16(
                    false, afrag[mi], false, bfrag[ni],
                    (short)0, acc[mi][ni], false, false);
    }

    // Epilogue: C/D layout -> VGPR r holds (M = r + 8*half_sel, N = lrow)
    #pragma unroll
    for (int mi = 0; mi < 2; ++mi) {
        #pragma unroll
        for (int ni = 0; ni < 4; ++ni) {
            const int r0 = block_row + wm * 32 + mi * 16 + half_sel * 8;
            const int c0 = block_col + wn * 64 + ni * 16 + lrow;
            #pragma unroll
            for (int r = 0; r < 8; ++r)
                C[(size_t)(r0 + r) * Ndim + c0] = acc[mi][ni][r];
        }
    }
}

// ---------------------------------------------------------------------------
extern "C" void kernel_launch(void* const* d_in, const int* in_sizes, int n_in,
                              void* d_out, int out_size, void* d_ws, size_t ws_size,
                              hipStream_t stream) {
    const float* x = (const float*)d_in[0];   // [2048, 2048]
    const float* w = (const float*)d_in[1];   // [2048, 2048]
    float* out = (float*)d_out;               // [2048, 2048]

    const float ALPHA_W = 1.0f, ALPHA_A = 0.9f;

    _Float16* xq = (_Float16*)d_ws;
    _Float16* wq = xq + (size_t)Mdim * Kdim;

    const int x_groups = (Mdim * Kdim) / GROUP;   // 32768
    const int w_groups = (Ndim * Kdim) / GROUP;   // 32768

    olive_quant_kernel<<<(x_groups + 7) / 8, 256, 0, stream>>>(x, xq, x_groups, 0.9f);
    olive_quant_kernel<<<(w_groups + 7) / 8, 256, 0, stream>>>(w, wq, w_groups, 1.0f);
    (void)ALPHA_W; (void)ALPHA_A;

    dim3 grid(Ndim / BN, Mdim / BM);
    olive_wmma_gemm<<<grid, 256, 0, stream>>>(xq, wq, out);
}
